// GCNet_18030272709117
// MI455X (gfx1250) — matrix operand order repaired and checked by reading, not compiled, not run on hardware
//
#include <hip/hip_runtime.h>

typedef __attribute__((ext_vector_type(16))) _Float16 v16h;
typedef __attribute__((ext_vector_type(8)))  float    v8f;

__device__ __forceinline__ void atomAdd(float* p, float v) {
    // native global_atomic_add_f32 (no CAS loop) -- atomics dominate this workload
    unsafeAtomicAdd(p, v);
}

// ---------------------------------------------------------------------------
// K1: per-node projections  hs = x @ W1p [N,256],  he = x @ W1e [N,128]
// ---------------------------------------------------------------------------
__global__ __launch_bounds__(384) void node_proj_kernel(
    const float* __restrict__ x, const float* __restrict__ W1p,
    const float* __restrict__ W1e, float* __restrict__ hs,
    float* __restrict__ he, int N)
{
    __shared__ float xr[5];
    const int n = blockIdx.x;
    const int f = threadIdx.x;
    if (f < 5) xr[f] = x[(size_t)n * 5 + f];
    __syncthreads();
    if (f < 256) {
        float acc = 0.f;
#pragma unroll
        for (int i = 0; i < 5; ++i) acc += xr[i] * W1p[i * 256 + f];
        hs[(size_t)n * 256 + f] = acc;
    } else {
        const int f2 = f - 256;
        float acc = 0.f;
#pragma unroll
        for (int i = 0; i < 5; ++i) acc += xr[i] * W1e[i * 128 + f2];
        he[(size_t)n * 128 + f2] = acc;
    }
}

// ---------------------------------------------------------------------------
// K2: edge scatter. 8 edges per 256-thread block, one 32-lane group per edge:
//     aggs[dst] += hs[src], agge[dst] += he[src], deg[src] += w, cnt[dst] += 1
// ---------------------------------------------------------------------------
__global__ __launch_bounds__(256) void edge_agg_kernel(
    const int* __restrict__ src, const int* __restrict__ dst,
    const float* __restrict__ w, const float* __restrict__ hs,
    const float* __restrict__ he, float* __restrict__ aggs,
    float* __restrict__ agge, float* __restrict__ deg,
    float* __restrict__ cnt, int E)
{
    const int lane = threadIdx.x & 31;
    const int sub  = threadIdx.x >> 5;
    const int e = blockIdx.x * 8 + sub;
    if (e >= E) return;
    const int s = src[e], d = dst[e];
    if (lane == 0) { atomAdd(&deg[s], w[e]); atomAdd(&cnt[d], 1.0f); }
    const float* __restrict__ hsp = hs + (size_t)s * 256;
    const float* __restrict__ hep = he + (size_t)s * 128;
    float* __restrict__ asp = aggs + (size_t)d * 256;
    float* __restrict__ aep = agge + (size_t)d * 128;
#pragma unroll
    for (int f = 0; f < 256; f += 32) atomAdd(&asp[f + lane], hsp[f + lane]);
#pragma unroll
    for (int f = 0; f < 128; f += 32) atomAdd(&aep[f + lane], hep[f + lane]);
}

// ---------------------------------------------------------------------------
// K3: mat_s = aggs/denom + x@R1p + b1p ; S = softmax(mat_s) (in place over
//     aggs; transposed f16 copy to ShT[256][Npad]); Y = relu(agge/denom +
//     x@R1e + b1e) (in place; transposed f16 copy to YhT[128][Npad]).
//     Also accumulates reg diag term sum(mask * S*S).
// ---------------------------------------------------------------------------
__global__ __launch_bounds__(256) void softmax_y_kernel(
    const float* __restrict__ x, const float* __restrict__ R1p,
    const float* __restrict__ b1p, const float* __restrict__ R1e,
    const float* __restrict__ b1e, const float* __restrict__ cnt,
    const float* __restrict__ deg, float* __restrict__ aggs,
    float* __restrict__ agge, _Float16* __restrict__ ShT,
    _Float16* __restrict__ YhT, float* __restrict__ regacc,
    int N, int Npad)
{
    __shared__ float xr[5];
    __shared__ float red[256];
    const int n = blockIdx.x;
    const int f = threadIdx.x;
    if (f < 5) xr[f] = x[(size_t)n * 5 + f];
    __syncthreads();
    const float dn = fmaxf(cnt[n], 1.0f);

    float v = aggs[(size_t)n * 256 + f] / dn + b1p[f];
#pragma unroll
    for (int i = 0; i < 5; ++i) v += xr[i] * R1p[i * 256 + f];

    // row max
    red[f] = v; __syncthreads();
    for (int s = 128; s > 0; s >>= 1) {
        if (f < s) red[f] = fmaxf(red[f], red[f + s]);
        __syncthreads();
    }
    const float mx = red[0]; __syncthreads();
    const float p = __expf(v - mx);
    red[f] = p; __syncthreads();
    for (int s = 128; s > 0; s >>= 1) {
        if (f < s) red[f] += red[f + s];
        __syncthreads();
    }
    const float Sv = p / red[0];
    __syncthreads();
    aggs[(size_t)n * 256 + f] = Sv;
    ShT[(size_t)f * Npad + n] = (_Float16)Sv;   // transposed: K(=node) contiguous

    if (f < 128) {
        float y = agge[(size_t)n * 128 + f] / dn + b1e[f];
#pragma unroll
        for (int i = 0; i < 5; ++i) y += xr[i] * R1e[i * 128 + f];
        y = fmaxf(y, 0.f);
        agge[(size_t)n * 128 + f] = y;
        YhT[(size_t)f * Npad + n] = (_Float16)y;
    }

    // diag reg term: mask(deg>0) * sum(S^2)
    red[f] = (deg[n] > 0.f) ? Sv * Sv : 0.f;
    __syncthreads();
    for (int s = 128; s > 0; s >>= 1) {
        if (f < s) red[f] += red[f + s];
        __syncthreads();
    }
    if (f == 0) atomAdd(&regacc[0], red[0]);
}

// ---------------------------------------------------------------------------
// K4: asn scatter (8 edges / block): asn_acc[src] += (w * d_is[dst]) * S[dst]
// ---------------------------------------------------------------------------
__global__ __launch_bounds__(256) void asn_scatter_kernel(
    const int* __restrict__ src, const int* __restrict__ dst,
    const float* __restrict__ w, const float* __restrict__ deg,
    const float* __restrict__ S, float* __restrict__ asn_acc, int E)
{
    const int lane = threadIdx.x & 31;
    const int sub  = threadIdx.x >> 5;
    const int e = blockIdx.x * 8 + sub;
    if (e >= E) return;
    const int s = src[e], d = dst[e];
    const float dg = deg[d];
    const float coef = (dg > 0.f) ? w[e] * rsqrtf(dg) : 0.f;
    if (coef != 0.f) {
        const float* __restrict__ Sp = S + (size_t)d * 256;
        float* __restrict__ ap = asn_acc + (size_t)s * 256;
#pragma unroll
        for (int f = 0; f < 256; f += 32) atomAdd(&ap[f + lane], coef * Sp[f + lane]);
    }
}

// ---------------------------------------------------------------------------
// K5: reg cross term: sum_n d_is[n] * dot(S[n], asn_acc[n])
// ---------------------------------------------------------------------------
__global__ __launch_bounds__(256) void reg_dot_kernel(
    const float* __restrict__ deg, const float* __restrict__ S,
    const float* __restrict__ asn_acc, float* __restrict__ regacc, int N)
{
    __shared__ float red[256];
    const int n = blockIdx.x, f = threadIdx.x;
    const float dg = deg[n];
    const float dis = (dg > 0.f) ? rsqrtf(dg) : 0.f;
    red[f] = dis * S[(size_t)n * 256 + f] * asn_acc[(size_t)n * 256 + f];
    __syncthreads();
    for (int s = 128; s > 0; s >>= 1) {
        if (f < s) red[f] += red[f + s];
        __syncthreads();
    }
    if (f == 0 && red[0] != 0.f) atomAdd(&regacc[1], red[0]);
}

// ---------------------------------------------------------------------------
// K6: pooled = S^T @ Y via v_wmma_f32_16x16x32_f16 on transposed f16 buffers.
//   ShT[256][Npad], YhT[128][Npad]; Npad multiple of 32, pad rows are zero,
//   so the K loop needs no bounds checks and EXEC stays all-ones.
//   Per step, per lane: 2x global_load_b128 for A + 2x for B.
//   A 16x32 layout: lane holds row M=l&15; K = {koff..koff+7, koff+16..koff+23},
//   koff = (lane>=16)?8:0.  B 32x16 layout: lane holds col N=l&15;
//   same K set as A: {koff..koff+7, koff+16..koff+23}.
// ---------------------------------------------------------------------------
__global__ __launch_bounds__(32) void pooled_wmma_kernel(
    const _Float16* __restrict__ ShT, const _Float16* __restrict__ YhT,
    float* __restrict__ pooled, int Npad)
{
    const int lane = threadIdx.x;             // 0..31
    const int b    = blockIdx.x;              // 128 tiles * 32 partitions
    const int part = b & 31;
    const int tile = b >> 5;                  // 0..127
    const int mt   = tile >> 3;               // 0..15 cluster tile
    const int kt   = tile & 7;                // 0..7  feature tile
    const int l15  = lane & 15;
    const bool hi  = lane >= 16;
    const int mrow = (mt << 4) + l15;         // A row  -> cluster index
    const int ncol = (kt << 4) + l15;         // B col  -> feature index
    const int koffA = hi ? 8 : 0;
    const int koffB = hi ? 8 : 0;
    const int nsteps = Npad >> 5;             // exact

    const _Float16* __restrict__ pA = ShT + (size_t)mrow * Npad + koffA;
    const _Float16* __restrict__ pB = YhT + (size_t)ncol * Npad + koffB;

    v8f c = {};
    for (int step = part; step < nsteps; step += 32) {
        const int n0 = step << 5;
        const int npf = n0 + 32 * 32;         // this wave's next chunk
        if (npf < Npad) {                     // uniform branch
            __builtin_prefetch(pA + npf, 0, 1);
            __builtin_prefetch(pB + npf, 0, 1);
        }
        union { uint4 q[2]; v16h v; } ua, ub;
        ua.q[0] = *(const uint4*)(pA + n0);        // K = koffA + 0..7
        ua.q[1] = *(const uint4*)(pA + n0 + 16);   // K = koffA + 16..23
        ub.q[0] = *(const uint4*)(pB + n0);        // K = koffB + 0..7
        ub.q[1] = *(const uint4*)(pB + n0 + 16);   // K = koffB + 16..23
        c = __builtin_amdgcn_wmma_f32_16x16x32_f16(
                false, ua.v, false, ub.v, (short)0, c, false, false);
    }

    // C/D layout: VGPR r -> M = r + (hi?8:0); N = l15
    const int mbase = (mt << 4) + (hi ? 8 : 0);
#pragma unroll
    for (int r = 0; r < 8; ++r)
        atomAdd(&pooled[(size_t)(mbase + r) * 128 + ncol], c[r]);
}

// ---------------------------------------------------------------------------
// K7: head. softmax over size-1 axis == 1 identically, so g = sum_m y2[m,:]
//     (W2p/R2p/b2p are dead). Then MLP + log_softmax + reg output.
// ---------------------------------------------------------------------------
__global__ __launch_bounds__(256) void head_kernel(
    const float* __restrict__ pooled, const float* __restrict__ W2e,
    const float* __restrict__ R2e, const float* __restrict__ b2e,
    const float* __restrict__ Wl1, const float* __restrict__ bl1,
    const float* __restrict__ Wl2, const float* __restrict__ bl2,
    const float* __restrict__ regacc, float* __restrict__ out, int N)
{
    __shared__ float mean_h[128];
    __shared__ float g[256];
    __shared__ float h1[256];
    __shared__ float logits[10];
    const int t = threadIdx.x;

    if (t < 128) {
        float acc = 0.f;
        for (int m = 0; m < 256; ++m) acc += pooled[m * 128 + t];
        mean_h[t] = acc * (1.0f / 256.0f);
    }
    __syncthreads();

    float c1t = b2e[t];
    for (int i = 0; i < 128; ++i) c1t += mean_h[i] * W2e[i * 256 + t];
    float gacc = 0.f;
    for (int m = 0; m < 256; ++m) {
        float acc = c1t;
        for (int i = 0; i < 128; ++i) acc += pooled[m * 128 + i] * R2e[i * 256 + t];
        gacc += fmaxf(acc, 0.f);
    }
    g[t] = gacc;
    __syncthreads();

    float h = bl1[t];
    for (int i = 0; i < 256; ++i) h += g[i] * Wl1[i * 256 + t];
    h1[t] = fmaxf(h, 0.f);
    __syncthreads();

    if (t < 10) {
        float acc = bl2[t];
        for (int i = 0; i < 256; ++i) acc += h1[i] * Wl2[i * 10 + t];
        logits[t] = acc;
    }
    __syncthreads();

    if (t == 0) {
        float mx = logits[0];
        for (int i = 1; i < 10; ++i) mx = fmaxf(mx, logits[i]);
        float s = 0.f;
        for (int i = 0; i < 10; ++i) s += __expf(logits[i] - mx);
        const float lse = mx + __logf(s);
        for (int i = 0; i < 10; ++i) out[i] = logits[i] - lse;
        out[10] = (regacc[0] - regacc[1]) / (float)N;
    }
}

// ---------------------------------------------------------------------------
extern "C" void kernel_launch(void* const* d_in, const int* in_sizes, int n_in,
                              void* d_out, int out_size, void* d_ws, size_t ws_size,
                              hipStream_t stream)
{
    const float* x   = (const float*)d_in[0];
    const int*   ei  = (const int*)d_in[1];
    const float* ew  = (const float*)d_in[2];
    const float* W1p = (const float*)d_in[3];
    const float* R1p = (const float*)d_in[4];
    const float* b1p = (const float*)d_in[5];
    const float* W1e = (const float*)d_in[6];
    const float* R1e = (const float*)d_in[7];
    const float* b1e = (const float*)d_in[8];
    // d_in[9..11] = W2p,R2p,b2p: provably unused (softmax over size-1 axis)
    const float* W2e = (const float*)d_in[12];
    const float* R2e = (const float*)d_in[13];
    const float* b2e = (const float*)d_in[14];
    const float* Wl1 = (const float*)d_in[15];
    const float* bl1 = (const float*)d_in[16];
    const float* Wl2 = (const float*)d_in[17];
    const float* bl2 = (const float*)d_in[18];

    const int N = in_sizes[0] / 5;
    const int E = in_sizes[2];
    const int* src = ei;
    const int* dst = ei + E;
    const int Npad = (N + 31) & ~31;          // K padding for guard-free WMMA
    const int Na   = (N + 63) & ~63;          // keep buffer starts 256B aligned

    // workspace layout (floats); every block size is a multiple of 64 floats
    float* p    = (float*)d_ws;
    float* deg  = p; p += Na;
    float* cnt  = p; p += Na;
    float* hs   = p; p += (size_t)N * 256;    // later reused as asn_acc
    float* he   = p; p += (size_t)N * 128;
    float* aggs = p; p += (size_t)N * 256;    // becomes S (f32)
    float* agge = p; p += (size_t)N * 128;    // becomes Y (f32)
    _Float16* ShT = (_Float16*)p; p += (size_t)Npad * 128;  // 256 x Npad halfs
    _Float16* YhT = (_Float16*)p; p += (size_t)Npad * 64;   // 128 x Npad halfs
    float* pooled = p; p += 256 * 128;
    float* regacc = p; p += 2;

    hipMemsetAsync(deg,    0, sizeof(float) * (size_t)2 * Na, stream);       // deg+cnt
    hipMemsetAsync(aggs,   0, sizeof(float) * (size_t)N * 384, stream);      // aggs+agge
    hipMemsetAsync(ShT,    0, sizeof(float) * (size_t)Npad * 192, stream);   // ShT+YhT (incl. pad)
    hipMemsetAsync(pooled, 0, sizeof(float) * (256 * 128 + 2), stream);      // pooled+regacc

    node_proj_kernel<<<N, 384, 0, stream>>>(x, W1p, W1e, hs, he, N);
    edge_agg_kernel<<<(E + 7) / 8, 256, 0, stream>>>(src, dst, ew, hs, he,
                                                     aggs, agge, deg, cnt, E);
    // hs free now -> reuse as asn_acc
    hipMemsetAsync(hs, 0, sizeof(float) * (size_t)N * 256, stream);
    softmax_y_kernel<<<N, 256, 0, stream>>>(x, R1p, b1p, R1e, b1e, cnt, deg,
                                            aggs, agge, ShT, YhT, regacc, N, Npad);
    asn_scatter_kernel<<<(E + 7) / 8, 256, 0, stream>>>(src, dst, ew, deg, aggs, hs, E);
    reg_dot_kernel<<<N, 256, 0, stream>>>(deg, aggs, hs, regacc, N);
    pooled_wmma_kernel<<<128 * 32, 32, 0, stream>>>(ShT, YhT, pooled, Npad);
    head_kernel<<<1, 256, 0, stream>>>(pooled, W2e, R2e, b2e, Wl1, bl1,
                                       Wl2, bl2, regacc, (float*)d_out, N);
}
